// LINet_6193342841713
// MI455X (gfx1250) — matrix operand order repaired and checked by reading, not compiled, not run on hardware
//
#include <hip/hip_runtime.h>
#include <hip/hip_bf16.h>
#include <math.h>

// ---------------- problem constants ----------------
#define Bg   128          // graphs
#define Nn   128          // nodes/graph
#define DEGc 8
#define Ec   (Bg*Nn*DEGc) // 131072 edges
#define INDc 16
#define Hc   32
#define K1c  64
#define K2c  32
#define OUTc 128
#define NCc  2
#define N0c  (Bg*Nn)      // 16384
#define N1c  (Bg*K1c)     // 8192
#define TPB  256

typedef _Float16 f16;
typedef __attribute__((ext_vector_type(16))) _Float16 v16h;
typedef __attribute__((ext_vector_type(8)))  _Float16 v8h;
typedef __attribute__((ext_vector_type(8)))  float    v8f;

// ---------------- WMMA fragment helpers (gfx1250, 16x16x32 f16) ----------------
// A is f16 (M x ldK) row-major, ldK multiple of 32, data zero-padded.
// Fragment layout: lane<16 holds K {0..15}, lane>=16 holds K {16..31}
// for row M = lane%16.  Two contiguous 16B loads per lane -> global_load_b128.
__device__ __forceinline__ v16h load_A_h(const f16* A, int row0, int ldK,
                                         int k0) {
  int lane = threadIdx.x & 31;
  int m = lane & 15, kh = lane >> 4;
  const f16* base = A + (size_t)(row0 + m) * ldK + k0 + kh * 16;
  v8h a0 = *(const v8h*)(base);
  v8h a1 = *(const v8h*)(base + 8);
  v16h a;
#pragma unroll
  for (int j = 0; j < 8; ++j) { a[j] = a0[j]; a[8 + j] = a1[j]; }
  return a;
}

// BT is f16 (N x ldK) row-major == B (KxN) transposed, zero-padded.
// Fragment: lane half kh holds K kh*16..kh*16+15 for col N = lane%16 ->
// one contiguous 32B load per lane.
__device__ __forceinline__ v16h load_B_h(const f16* BT, int col0, int ldK,
                                         int k0) {
  int lane = threadIdx.x & 31;
  int c = lane & 15, kh = lane >> 4;
  return *(const v16h*)(BT + (size_t)(col0 + c) * ldK + k0 + kh * 16);
}

// D: VGPR r, lane L -> row = 8*(L/16)+r, col = L%16. Unconditional stores.
__device__ __forceinline__ void store_D(float* dst, v8f d, int row0, int col0,
                                        int ld) {
  int lane  = threadIdx.x & 31;
  int c     = col0 + (lane & 15);
  int rbase = row0 + ((lane >> 4) * 8);
#pragma unroll
  for (int r = 0; r < 8; ++r) dst[(size_t)(rbase + r) * ld + c] = d[r];
}

__device__ __forceinline__ v8f wmma_f16(v16h a, v16h b, v8f c) {
  return __builtin_amdgcn_wmma_f32_16x16x32_f16(false, a, false, b, (short)0, c,
                                                false, false);
}

// ---------------- precision/layout conversion kernels ----------------
// f32 (M x Ksrc) row-major -> f16 (M x Kpad) row-major, zero-padded K.
__global__ void k_cvt_pad(f16* dst, const float* src, int M, int Ksrc,
                          int Kpad) {
  int i = blockIdx.x * blockDim.x + threadIdx.x;
  if (i >= M * Kpad) return;
  int r = i / Kpad, k = i - r * Kpad;
  dst[i] = (k < Ksrc) ? (f16)src[r * Ksrc + k] : (f16)0.0f;
}

// f32 weights (K x N) row-major -> f16 (N x Kpad) transposed, zero-padded.
__global__ void k_cvt_wT(f16* dst, const float* src, int K, int N, int Kpad) {
  int i = blockIdx.x * blockDim.x + threadIdx.x;
  if (i >= N * Kpad) return;
  int n = i / Kpad, k = i - n * Kpad;
  dst[i] = (k < K) ? (f16)src[k * N + n] : (f16)0.0f;
}

// ---------------- compute kernels ----------------

// t = relu(e*w1 + b1) for both NNConv edge-MLPs
__global__ void k_edge_mlp(float* t1, float* t2, const float* e,
                           const float* w1a, const float* b1a,
                           const float* w1b, const float* b1b) {
  int i = blockIdx.x * blockDim.x + threadIdx.x;
  if (i >= Ec) return;
  float ev = e[i];
  t1[i] = fmaxf(ev * w1a[0] + b1a[0], 0.0f);
  t2[i] = fmaxf(ev * w1b[0] + b1b[0], 0.0f);
}

// U = A @ W2 (Mx32 * 32x32), V = A @ B2 ; one wave per 16-row block; 4 WMMAs.
__global__ void k_node_gemm(float* U, float* V, const f16* A, const f16* WT,
                            const f16* BT, int M) {
  int wave = (blockIdx.x * blockDim.x + threadIdx.x) >> 5;
  int row0 = wave * 16;
  if (row0 >= M) return;  // wave-uniform (exact grids anyway)
  v16h a   = load_A_h(A, row0, 32, 0);
  v16h bw0 = load_B_h(WT, 0, 32, 0);
  v16h bw1 = load_B_h(WT, 16, 32, 0);
  v16h bb0 = load_B_h(BT, 0, 32, 0);
  v16h bb1 = load_B_h(BT, 16, 32, 0);
  v8f c0 = {}, c1 = {}, c2 = {}, c3 = {};
  c0 = wmma_f16(a, bw0, c0);
  c1 = wmma_f16(a, bw1, c1);
  c2 = wmma_f16(a, bb0, c2);
  c3 = wmma_f16(a, bb1, c3);
  store_D(U, c0, row0, 0, Hc);
  store_D(U, c1, row0, 16, Hc);
  store_D(V, c2, row0, 0, Hc);
  store_D(V, c3, row0, 16, Hc);
}

// C(MxN,f32) = A(MxKpad,f16) @ BT(NxKpad,f16)^T ; one wave per 16x16 tile.
__global__ void k_fc_gemm(float* C, const f16* A, const f16* BT, int M,
                          int Kpad, int N) {
  int wave   = (blockIdx.x * blockDim.x + threadIdx.x) >> 5;
  int tilesN = N >> 4;
  int tiles  = (M >> 4) * tilesN;
  if (wave >= tiles) return;  // wave-uniform
  int tm = wave / tilesN, tn = wave - tm * tilesN;
  v8f c = {};
  for (int k0 = 0; k0 < Kpad; k0 += 32) {
    v16h a = load_A_h(A, tm * 16, Kpad, k0);
    v16h b = load_B_h(BT, tn * 16, Kpad, k0);
    c = wmma_f16(a, b, c);
  }
  store_D(C, c, tm * 16, tn * 16, N);
}

// initialize accumulator rows with bias vector (H=32 columns)
__global__ void k_fill_bias(float* dst, const float* bias, int total) {
  int i = blockIdx.x * blockDim.x + threadIdx.x;
  if (i >= total) return;
  dst[i] = bias[i & (Hc - 1)];
}

__global__ void k_zero(float* dst, int n) {
  int i = blockIdx.x * blockDim.x + threadIdx.x;
  if (i < n) dst[i] = 0.0f;
}

// msg scatter: acc[dst] += valid * (t*U[src] + V[src]); thread per edge-feature
__global__ void k_scatter_msg(float* acc, const float* U, const float* V,
                              const float* t, const int* src, const int* dst,
                              const float* valid) {
  int i = blockIdx.x * blockDim.x + threadIdx.x;
  if (i >= Ec * Hc) return;
  int e = i >> 5, f = i & (Hc - 1);
  float v = valid ? valid[e] : 1.0f;
  if (v == 0.0f) return;
  int s = src[e];
  float m = (t[e] * U[s * Hc + f] + V[s * Hc + f]) * v;
  atomicAdd(&acc[dst[e] * Hc + f], m);
}

__global__ void k_degrees(float* od, float* id_, const int* src,
                          const int* dst, const float* valid) {
  int e = blockIdx.x * blockDim.x + threadIdx.x;
  if (e >= Ec) return;
  float v = valid ? valid[e] : 1.0f;
  atomicAdd(&od[src[e]], v);
  atomicAdd(&id_[dst[e]], v);
}

// hw[n] = dot(h[n], w)  (GCN projection to scalar score channel)
__global__ void k_gcn_gather(float* hw, const float* hx, const float* w,
                             int n) {
  int i = blockIdx.x * blockDim.x + threadIdx.x;
  if (i >= n) return;
  float s = 0.0f;
#pragma unroll
  for (int f = 0; f < Hc; ++f) s += hx[i * Hc + f] * w[f];
  hw[i] = s;
}

__global__ void k_gcn_scatter(float* agg, const float* hw, const int* src,
                              const int* dst, const float* valid,
                              const float* od) {
  int e = blockIdx.x * blockDim.x + threadIdx.x;
  if (e >= Ec) return;
  float v = valid ? valid[e] : 1.0f;
  if (v == 0.0f) return;
  int s = src[e];
  float m = hw[s] * v * rsqrtf(fmaxf(od[s], 1.0f));
  atomicAdd(&agg[dst[e]], m);
}

__global__ void k_gcn_score(float* score, const float* agg, const float* id_,
                            const float* bias, int n) {
  int i = blockIdx.x * blockDim.x + threadIdx.x;
  if (i >= n) return;
  float z = agg[i] * rsqrtf(fmaxf(id_[i], 1.0f)) + bias[0];
  score[i] = 1.0f / (1.0f + expf(-z));
}

// per-graph top-k (descending, lowest-index ties first), gate + relabel.
__global__ void k_topk_pool(float* hnew, float* kept, int* relabel,
                            const float* score, const float* hold, int nper,
                            int k) {
  __shared__ float s[128];
  int b = blockIdx.x, i = threadIdx.x;
  int g0 = b * nper;
  s[i] = score[g0 + i];
  __syncthreads();
  float mys = s[i];
  int rank = 0;
  for (int j = 0; j < nper; ++j) {
    float sj = s[j];
    rank += (sj > mys) || (sj == mys && j < i);
  }
  int old  = g0 + i;
  int keep = rank < k;
  kept[old]    = keep ? 1.0f : 0.0f;
  int nid      = keep ? (b * k + rank) : 0;
  relabel[old] = nid;
  if (keep) {
#pragma unroll
    for (int f = 0; f < Hc; ++f)
      hnew[nid * Hc + f] = hold[old * Hc + f] * mys;
  }
}

__global__ void k_relabel(int* ns, int* nd, float* nv, const int* src,
                          const int* dst, const float* kept,
                          const int* relabel, const float* valid) {
  int e = blockIdx.x * blockDim.x + threadIdx.x;
  if (e >= Ec) return;
  int s = src[e], d = dst[e];
  ns[e] = relabel[s];
  nd[e] = relabel[d];
  float v = valid ? valid[e] : 1.0f;
  nv[e] = v * kept[s] * kept[d];
}

// per-graph readout: mean || max over k kept nodes
__global__ void k_readout(float* x, int coloff, const float* hp, int k) {
  int b = blockIdx.x, f = threadIdx.x;  // Hc threads
  const float* base = hp + (size_t)b * k * Hc + f;
  float sum = 0.0f, mx = -INFINITY;
  for (int n = 0; n < k; ++n) {
    float v = base[n * Hc];
    sum += v;
    mx = fmaxf(mx, v);
  }
  x[b * (4 * Hc) + coloff + f]      = sum / (float)k;
  x[b * (4 * Hc) + coloff + Hc + f] = mx;
}

// y = relu(x + bias) * (1/sqrt(1+eps)) * g + bb   (eval-mode BN, init stats)
__global__ void k_post_bn(float* y, const float* x, const float* bias,
                          const float* g, const float* bb, int rows,
                          int cols) {
  int i = blockIdx.x * blockDim.x + threadIdx.x;
  if (i >= rows * cols) return;
  int c = i % cols;
  const float inv = rsqrtf(1.0f + 1e-5f);
  float v = fmaxf(x[i] + bias[c], 0.0f);
  y[i] = v * inv * g[c] + bb[c];
}

// final 32->2 classifier + log_softmax
__global__ void k_fc3_head(float* out, const float* y2, const float* w,
                           const float* b) {
  int r = blockIdx.x * blockDim.x + threadIdx.x;
  if (r >= Bg) return;
  float z0 = b[0], z1 = b[1];
#pragma unroll
  for (int f = 0; f < Hc; ++f) {
    float v = y2[r * Hc + f];
    z0 += v * w[f * 2 + 0];
    z1 += v * w[f * 2 + 1];
  }
  float m = fmaxf(z0, z1);
  float l = m + logf(expf(z0 - m) + expf(z1 - m));
  out[r * 2 + 0] = z0 - l;
  out[r * 2 + 1] = z1 - l;
}

// ---------------- host orchestration ----------------
static inline int cdiv(int a, int b) { return (a + b - 1) / b; }

extern "C" void kernel_launch(void* const* d_in, const int* in_sizes, int n_in,
                              void* d_out, int out_size, void* d_ws,
                              size_t ws_size, hipStream_t stream) {
  const float* h_in   = (const float*)d_in[0];
  const float* e_in   = (const float*)d_in[1];
  const int*   src    = (const int*)d_in[2];
  const int*   dst    = (const int*)d_in[3];
  const float* n1_w1  = (const float*)d_in[4];
  const float* n1_b1  = (const float*)d_in[5];
  const float* n1_w2  = (const float*)d_in[6];
  const float* n1_b2  = (const float*)d_in[7];
  const float* conv1b = (const float*)d_in[8];
  const float* p1_w   = (const float*)d_in[9];
  const float* p1_b   = (const float*)d_in[10];
  const float* n2_w1  = (const float*)d_in[11];
  const float* n2_b1  = (const float*)d_in[12];
  const float* n2_w2  = (const float*)d_in[13];
  const float* n2_b2  = (const float*)d_in[14];
  const float* conv2b = (const float*)d_in[15];
  const float* p2_w   = (const float*)d_in[16];
  const float* p2_b   = (const float*)d_in[17];
  const float* fc1_w  = (const float*)d_in[18];
  const float* fc1_b  = (const float*)d_in[19];
  const float* bn4_g  = (const float*)d_in[20];
  const float* bn4_b  = (const float*)d_in[21];
  const float* fc2_w  = (const float*)d_in[22];
  const float* fc2_b  = (const float*)d_in[23];
  const float* bn5_g  = (const float*)d_in[24];
  const float* bn5_b  = (const float*)d_in[25];
  const float* fc3_w  = (const float*)d_in[26];
  const float* fc3_b  = (const float*)d_in[27];

  float* out    = (float*)d_out;   // x: 256 floats
  float* score1 = out + Bg * NCc;  // 16384
  float* score2 = score1 + N0c;    // 8192

  // carve workspace (4-byte units; f16 buffers consume ceil(n/2) units)
  float* W = (float*)d_ws;
  size_t o = 0;
  auto F = [&](size_t n) { float* p = W + o; o += n; return p; };
  auto Hbuf = [&](size_t n) { f16* p = (f16*)(W + o); o += (n + 1) / 2; return p; };

  float* t1    = F(Ec);
  float* t2    = F(Ec);
  float* u1    = F((size_t)N0c * Hc);
  float* v1n   = F((size_t)N0c * Hc);
  float* h1    = F((size_t)N0c * Hc);
  float* h1p   = F((size_t)N1c * Hc);
  float* u2    = F((size_t)N1c * Hc);
  float* v2n   = F((size_t)N1c * Hc);
  float* h2    = F((size_t)N1c * Hc);
  float* h2p   = F((size_t)Bg * K2c * Hc);
  float* dego1 = F(N0c); float* degi1 = F(N0c);
  float* agg1  = F(N0c); float* hw1   = F(N0c);
  float* dego2 = F(N1c); float* degi2 = F(N1c);
  float* agg2  = F(N1c); float* hw2   = F(N1c);
  float* kept1 = F(N0c); float* kept2 = F(N1c);
  int*   rel1  = (int*)F(N0c);
  int*   rel2  = (int*)F(N1c);
  int*   s1    = (int*)F(Ec);
  int*   d1    = (int*)F(Ec);
  float* v1    = F(Ec);
  float* xbuf  = F((size_t)Bg * 4 * Hc);
  float* y1r   = F((size_t)Bg * OUTc);
  float* y1    = F((size_t)Bg * OUTc);
  float* y2r   = F((size_t)Bg * Hc);
  float* y2    = F((size_t)Bg * Hc);
  // f16 staging buffers (padded / transposed layouts for WMMA)
  f16* hA    = Hbuf((size_t)N0c * 32);   // h zero-padded K 16->32
  f16* h1pA  = Hbuf((size_t)N1c * 32);
  f16* xA    = Hbuf((size_t)Bg * 128);
  f16* y1A   = Hbuf((size_t)Bg * 128);
  f16* w2aT  = Hbuf(32 * 32);            // (N=32 x Kpad=32)
  f16* b2aT  = Hbuf(32 * 32);
  f16* w2bT  = Hbuf(32 * 32);
  f16* b2bT  = Hbuf(32 * 32);
  f16* fc1T  = Hbuf(128 * 128);          // (N=128 x K=128)
  f16* fc2T  = Hbuf(32 * 128);           // (N=32  x K=128)
  (void)ws_size; (void)n_in; (void)in_sizes; (void)out_size;

  // 0) precision/layout staging for WMMA operands
  k_cvt_pad<<<cdiv(N0c * 32, TPB), TPB, 0, stream>>>(hA, h_in, N0c, INDc, 32);
  k_cvt_wT<<<cdiv(32 * 32, TPB), TPB, 0, stream>>>(w2aT, n1_w2, INDc, Hc, 32);
  k_cvt_wT<<<cdiv(32 * 32, TPB), TPB, 0, stream>>>(b2aT, n1_b2, INDc, Hc, 32);
  k_cvt_wT<<<cdiv(32 * 32, TPB), TPB, 0, stream>>>(w2bT, n2_w2, Hc, Hc, 32);
  k_cvt_wT<<<cdiv(32 * 32, TPB), TPB, 0, stream>>>(b2bT, n2_b2, Hc, Hc, 32);
  k_cvt_wT<<<cdiv(128 * 128, TPB), TPB, 0, stream>>>(fc1T, fc1_w, 128, 128, 128);
  k_cvt_wT<<<cdiv(32 * 128, TPB), TPB, 0, stream>>>(fc2T, fc2_w, 128, 32, 128);

  // 1) edge MLP scalars (both layers)
  k_edge_mlp<<<cdiv(Ec, TPB), TPB, 0, stream>>>(t1, t2, e_in, n1_w1, n1_b1,
                                                n2_w1, n2_b1);
  // 2) node GEMM layer1: u1 = h@W2, v1n = h@B2  [WMMA]
  k_node_gemm<<<(N0c / 16) / 8, TPB, 0, stream>>>(u1, v1n, hA, w2aT, b2aT, N0c);
  // 3) NNConv1 aggregate
  k_fill_bias<<<cdiv(N0c * Hc, TPB), TPB, 0, stream>>>(h1, conv1b, N0c * Hc);
  k_scatter_msg<<<cdiv(Ec * Hc, TPB), TPB, 0, stream>>>(h1, u1, v1n, t1, src,
                                                        dst, nullptr);
  // 4) SAGPool1 GCN score
  k_zero<<<cdiv(N0c, TPB), TPB, 0, stream>>>(dego1, N0c);
  k_zero<<<cdiv(N0c, TPB), TPB, 0, stream>>>(degi1, N0c);
  k_zero<<<cdiv(N0c, TPB), TPB, 0, stream>>>(agg1, N0c);
  k_degrees<<<cdiv(Ec, TPB), TPB, 0, stream>>>(dego1, degi1, src, dst, nullptr);
  k_gcn_gather<<<cdiv(N0c, TPB), TPB, 0, stream>>>(hw1, h1, p1_w, N0c);
  k_gcn_scatter<<<cdiv(Ec, TPB), TPB, 0, stream>>>(agg1, hw1, src, dst, nullptr,
                                                   dego1);
  k_gcn_score<<<cdiv(N0c, TPB), TPB, 0, stream>>>(score1, agg1, degi1, p1_b,
                                                  N0c);
  // 5) top-k pool 1 + relabel
  k_topk_pool<<<Bg, Nn, 0, stream>>>(h1p, kept1, rel1, score1, h1, Nn, K1c);
  k_relabel<<<cdiv(Ec, TPB), TPB, 0, stream>>>(s1, d1, v1, src, dst, kept1,
                                               rel1, nullptr);
  // 6) readout g1 -> x[:, 0:64]
  k_readout<<<Bg, Hc, 0, stream>>>(xbuf, 0, h1p, K1c);
  // 7) node GEMM layer2 [WMMA]
  k_cvt_pad<<<cdiv(N1c * 32, TPB), TPB, 0, stream>>>(h1pA, h1p, N1c, Hc, 32);
  k_node_gemm<<<(N1c / 16) / 8, TPB, 0, stream>>>(u2, v2n, h1pA, w2bT, b2bT,
                                                  N1c);
  // 8) NNConv2 aggregate (masked)
  k_fill_bias<<<cdiv(N1c * Hc, TPB), TPB, 0, stream>>>(h2, conv2b, N1c * Hc);
  k_scatter_msg<<<cdiv(Ec * Hc, TPB), TPB, 0, stream>>>(h2, u2, v2n, t2, s1,
                                                        d1, v1);
  // 9) SAGPool2 score
  k_zero<<<cdiv(N1c, TPB), TPB, 0, stream>>>(dego2, N1c);
  k_zero<<<cdiv(N1c, TPB), TPB, 0, stream>>>(degi2, N1c);
  k_zero<<<cdiv(N1c, TPB), TPB, 0, stream>>>(agg2, N1c);
  k_degrees<<<cdiv(Ec, TPB), TPB, 0, stream>>>(dego2, degi2, s1, d1, v1);
  k_gcn_gather<<<cdiv(N1c, TPB), TPB, 0, stream>>>(hw2, h2, p2_w, N1c);
  k_gcn_scatter<<<cdiv(Ec, TPB), TPB, 0, stream>>>(agg2, hw2, s1, d1, v1,
                                                   dego2);
  k_gcn_score<<<cdiv(N1c, TPB), TPB, 0, stream>>>(score2, agg2, degi2, p2_b,
                                                  N1c);
  // 10) top-k pool 2 + readout g2 -> x[:, 64:128]
  k_topk_pool<<<Bg, K1c, 0, stream>>>(h2p, kept2, rel2, score2, h2, K1c, K2c);
  k_readout<<<Bg, Hc, 0, stream>>>(xbuf, 2 * Hc, h2p, K2c);
  // 11) FC head [WMMA]
  k_cvt_pad<<<cdiv(Bg * 128, TPB), TPB, 0, stream>>>(xA, xbuf, Bg, 128, 128);
  k_fc_gemm<<<8, TPB, 0, stream>>>(y1r, xA, fc1T, Bg, 128, OUTc);  // 64 tiles
  k_post_bn<<<cdiv(Bg * OUTc, TPB), TPB, 0, stream>>>(y1, y1r, fc1_b, bn4_g,
                                                      bn4_b, Bg, OUTc);
  k_cvt_pad<<<cdiv(Bg * 128, TPB), TPB, 0, stream>>>(y1A, y1, Bg, 128, 128);
  k_fc_gemm<<<2, TPB, 0, stream>>>(y2r, y1A, fc2T, Bg, 128, Hc);   // 16 tiles
  k_post_bn<<<cdiv(Bg * Hc, TPB), TPB, 0, stream>>>(y2, y2r, fc2_b, bn5_g,
                                                    bn5_b, Bg, Hc);
  k_fc3_head<<<cdiv(Bg, TPB), TPB, 0, stream>>>(out, y2, fc3_w, fc3_b);
}
